// GNNEdgeClassifier_73882027426425
// MI455X (gfx1250) — matrix operand order repaired and checked by reading, not compiled, not run on hardware
//
#include <hip/hip_runtime.h>
#include <hip/hip_bf16.h>
#include <math.h>

// ---------------------------------------------------------------------------
// Types for WMMA (CDNA5 / gfx1250, wave32)
// ---------------------------------------------------------------------------
typedef __attribute__((ext_vector_type(16))) __bf16 v16bf;
typedef __attribute__((ext_vector_type(8)))  __bf16 v8bf;
typedef __attribute__((ext_vector_type(8)))  float  v8f;

#define HID 128
#define NCONV 4
#define CROWS 64   // conv rows per block (4 M sub-tiles per wave)
#define QROWS 32   // MLP query edges per block (2 M sub-tiles per wave)

// ---------------------------------------------------------------------------
// WMMA fragment loaders (bf16, 16x16x32)
// A matrix 16x32 (MxK): lane m = lane&15; lane group (lane>>4) selects
//   kb = 0 or 16; half-elements i -> K = kb+i.
//   => two contiguous 8x-bf16 (16B) loads at +kb and +kb+8.
// B matrix 32x16 (KxN), stored transposed (n-major, ld = K-extent):
//   lane n = lane&15; lanes 0-15 hold K=0..15, lanes 16-31 hold K=16..31
//   => 16 contiguous bf16 (two 16B loads) at +kb16.
// ---------------------------------------------------------------------------
__device__ __forceinline__ v16bf frag_a(const __bf16* base, int row, int ld,
                                        int k0, int lane) {
  const int kb = (lane >> 4) * 16;
  const __bf16* p = base + row * ld + k0 + kb;
  v8bf lo = *(const v8bf*)(p);
  v8bf hi = *(const v8bf*)(p + 8);
  v16bf a;
#pragma unroll
  for (int i = 0; i < 8; ++i) { a[i] = lo[i]; a[i + 8] = hi[i]; }
  return a;
}

__device__ __forceinline__ v16bf frag_b(const __bf16* baseT, int col, int ld,
                                        int k0, int lane) {
  const int kb = (lane >> 4) * 16;
  const __bf16* p = baseT + col * ld + k0 + kb;
  v8bf lo = *(const v8bf*)(p);
  v8bf hi = *(const v8bf*)(p + 8);
  v16bf b;
#pragma unroll
  for (int i = 0; i < 8; ++i) { b[i] = lo[i]; b[i + 8] = hi[i]; }
  return b;
}

__device__ __forceinline__ float gelu_exact(float v) {
  return 0.5f * v * (1.f + erff(v * 0.70710678118654752f));
}

// ---------------------------------------------------------------------------
// Workspace zero fill
// ---------------------------------------------------------------------------
__global__ void zero_f32(float* __restrict__ p, size_t n) {
  size_t i = (size_t)blockIdx.x * blockDim.x + threadIdx.x;
  size_t stride = (size_t)gridDim.x * blockDim.x;
  for (; i < n; i += stride) p[i] = 0.f;
}

// ---------------------------------------------------------------------------
// One-time: convert + transpose weights to bf16 (L2-resident, ~320KB total)
//   relwt/rootwt[l][n*128 + k] = w[l][k*128 + n]
//   w1t[n*256 + k]             = w1[k*128 + n]
// ---------------------------------------------------------------------------
__global__ void prep_weights(const float* __restrict__ relw,
                             const float* __restrict__ rootw,
                             const float* __restrict__ w1,
                             __bf16* __restrict__ relwt,
                             __bf16* __restrict__ rootwt,
                             __bf16* __restrict__ w1t) {
  int idx = blockIdx.x * blockDim.x + threadIdx.x;
  const int nconv = NCONV * HID * HID;
  if (idx < nconv) {
    int l = idx >> 14;
    int r = idx & 16383;
    int k = r >> 7;
    int n = r & 127;
    int dst = (l << 14) + (n << 7) + k;
    relwt[dst]  = (__bf16)relw[idx];
    rootwt[dst] = (__bf16)rootw[idx];
  }
  if (idx < 2 * HID * HID) {  // 256 x 128
    int k = idx >> 7;
    int n = idx & 127;
    w1t[n * 256 + k] = (__bf16)w1[idx];
  }
}

// ---------------------------------------------------------------------------
// Scatter: agg[dst] += x[src] * edge_attr   (one wave32 per edge iteration,
// 4 channels/lane). Node features (51MB) are L2-resident. Warp-stride loop
// + global_prefetch_b8 of the next iteration's source row hides L2 gather
// latency behind the f32 atomic issue of the current edge.
// ---------------------------------------------------------------------------
__global__ void scatter_edges(const float* __restrict__ x,
                              const int* __restrict__ ei,
                              const float* __restrict__ ea,
                              float* __restrict__ agg, int E) {
  const int lane = threadIdx.x & 31;
  const int warp = (blockIdx.x * blockDim.x + threadIdx.x) >> 5;
  const int nwarps = (gridDim.x * blockDim.x) >> 5;
  for (int e = warp; e < E; e += nwarps) {
    int en = e + nwarps;
    if (en < E) {
      int sn = ei[en];
      __builtin_prefetch(x + (size_t)sn * HID + lane * 4, 0, 1);
    }
    int s = ei[e];
    int d = ei[E + e];
    float w = ea[e];
    const float4 v = *(const float4*)(x + (size_t)s * HID + lane * 4);
    float* p = agg + (size_t)d * HID + lane * 4;
    atomicAdd(p + 0, v.x * w);
    atomicAdd(p + 1, v.y * w);
    atomicAdd(p + 2, v.z * w);
    atomicAdd(p + 3, v.w * w);
  }
}

// ---------------------------------------------------------------------------
// h = GELU( agg @ rel_w + rel_b + x @ root_w )
// Block: 256 threads = 8 waves; 64 output rows x 128 cols per block.
// Each wave: 4 M sub-tiles of 16x16; B fragments (rel & root) loaded once per
// k-step and reused across the 4 sub-tiles -> 32 WMMAs/wave, 4x less B L2
// traffic than a 16-row tile. Activations staged f32->bf16 in LDS.
// ---------------------------------------------------------------------------
__global__ void conv_gemm_gelu(const float* __restrict__ agg,
                               const float* __restrict__ xin,
                               const __bf16* __restrict__ relwt,
                               const __bf16* __restrict__ rootwt,
                               const float* __restrict__ relb,
                               float* __restrict__ h, int Nn) {
  __shared__ __bf16 lAgg[CROWS * HID];  // 16KB
  __shared__ __bf16 lX[CROWS * HID];    // 16KB
  const int t = threadIdx.x;
  const int row0 = blockIdx.x * CROWS;

  // Stage CROWS rows of agg and x as bf16 (zero-pad past N).
  for (int i = t; i < CROWS * HID; i += 256) {
    int r = i >> 7;
    int c = i & 127;
    int gr = row0 + r;
    float va = 0.f, vx = 0.f;
    if (gr < Nn) {
      va = agg[(size_t)gr * HID + c];
      vx = xin[(size_t)gr * HID + c];
    }
    lAgg[i] = (__bf16)va;
    lX[i]   = (__bf16)vx;
  }
  __syncthreads();

  const int lane = t & 31;
  const int ncol = (t >> 5) * 16 + (lane & 15);
  const int m = lane & 15;

  v8f acc[4];
  const float bias = relb[ncol];
#pragma unroll
  for (int s = 0; s < 4; ++s)
#pragma unroll
    for (int r = 0; r < 8; ++r) acc[s][r] = bias;

#pragma unroll
  for (int k0 = 0; k0 < HID; k0 += 32) {
    v16bf bR = frag_b(relwt, ncol, HID, k0, lane);
    v16bf bT = frag_b(rootwt, ncol, HID, k0, lane);
#pragma unroll
    for (int s = 0; s < 4; ++s) {
      v16bf aA = frag_a(lAgg + s * 16 * HID, m, HID, k0, lane);
      acc[s] = __builtin_amdgcn_wmma_f32_16x16x32_bf16(false, aA, false, bR,
                                                       (short)0, acc[s],
                                                       false, false);
      v16bf aX = frag_a(lX + s * 16 * HID, m, HID, k0, lane);
      acc[s] = __builtin_amdgcn_wmma_f32_16x16x32_bf16(false, aX, false, bT,
                                                       (short)0, acc[s],
                                                       false, false);
    }
  }

  // GELU + store. D layout: VGPR r -> row = r + (lane>>4)*8, col = ncol.
#pragma unroll
  for (int s = 0; s < 4; ++s) {
#pragma unroll
    for (int r = 0; r < 8; ++r) {
      int grow = row0 + s * 16 + r + ((lane >> 4) * 8);
      if (grow < Nn) h[(size_t)grow * HID + ncol] = gelu_exact(acc[s][r]);
    }
  }
}

// ---------------------------------------------------------------------------
// BatchNorm stats: stats[0:128]=sum, stats[128:256]=sum of squares.
// ---------------------------------------------------------------------------
__global__ void bn_stats(const float* __restrict__ h, float* __restrict__ stats,
                         int Nn) {
  __shared__ float ls[256];
  __shared__ float ls2[256];
  const int t = threadIdx.x;
  const int c = t & 127;
  const int half = t >> 7;
  float s = 0.f, s2 = 0.f;
  for (int r = blockIdx.x * 2 + half; r < Nn; r += gridDim.x * 2) {
    float v = h[(size_t)r * HID + c];
    s += v;
    s2 += v * v;
  }
  ls[t] = s;
  ls2[t] = s2;
  __syncthreads();
  if (t < 128) {
    atomicAdd(&stats[c], ls[t] + ls[t + 128]);
    atomicAdd(&stats[128 + c], ls2[t] + ls2[t + 128]);
  }
}

// ---------------------------------------------------------------------------
// BatchNorm apply: xout = (h - mean) * rsqrt(var+eps) * g + b
// ---------------------------------------------------------------------------
__global__ void bn_apply(const float* __restrict__ h,
                         const float* __restrict__ stats,
                         const float* __restrict__ g,
                         const float* __restrict__ b,
                         float* __restrict__ xout, int Nn) {
  __shared__ float scale[128];
  __shared__ float shift[128];
  const int t = threadIdx.x;
  if (t < 128) {
    float inv = 1.f / (float)Nn;
    float mean = stats[t] * inv;
    float var = stats[128 + t] * inv - mean * mean;
    float sc = rsqrtf(var + 1e-5f) * g[t];
    scale[t] = sc;
    shift[t] = b[t] - mean * sc;
  }
  __syncthreads();
  size_t total = (size_t)Nn * HID;
  size_t stride = (size_t)gridDim.x * blockDim.x;
  for (size_t i = (size_t)blockIdx.x * blockDim.x + t; i < total; i += stride) {
    int ch = (int)(i & 127);
    xout[i] = h[i] * scale[ch] + shift[ch];
  }
}

// ---------------------------------------------------------------------------
// Edge MLP: out = sigmoid( GELU([x[s],x[d]] @ W1 + b1) @ w2 + b2 )
// Block: 32 query edges; ef tile 32x256 bf16 in LDS; 8 waves, each owning 16
// cols with 2 M sub-tiles; W1 B-fragments loaded once per k-step and reused
// -> 16 WMMAs/wave, half the W1 L2 traffic of a 16-edge tile.
// ---------------------------------------------------------------------------
__global__ void edge_mlp(const float* __restrict__ x,
                         const int* __restrict__ eli,
                         const __bf16* __restrict__ w1t,
                         const float* __restrict__ b1,
                         const float* __restrict__ w2,
                         const float* __restrict__ b2,
                         float* __restrict__ out, int EQ) {
  __shared__ __bf16 ef[QROWS * 256];   // 16KB
  __shared__ float hmid[QROWS * HID];  // 16KB
  const int t = threadIdx.x;
  const int q0 = blockIdx.x * QROWS;

  // Stage concatenated features (bf16): thread t owns edge t>>3, 32 features.
  {
    int e = t >> 3;       // local edge 0..31
    int fg = (t & 7) * 32;  // feature base
    int q = q0 + e;
    bool valid = q < EQ;
    int sidx = 0, didx = 0;
    if (valid) {
      sidx = eli[q];
      didx = eli[EQ + q];
    }
#pragma unroll
    for (int i = 0; i < 32; ++i) {
      int f = fg + i;
      float v = 0.f;
      if (valid)
        v = (f < HID) ? x[(size_t)sidx * HID + f]
                      : x[(size_t)didx * HID + (f - HID)];
      ef[e * 256 + f] = (__bf16)v;
    }
  }
  __syncthreads();

  const int lane = t & 31;
  const int ncol = (t >> 5) * 16 + (lane & 15);
  const int m = lane & 15;

  v8f acc[2];
  const float bias = b1[ncol];
#pragma unroll
  for (int s = 0; s < 2; ++s)
#pragma unroll
    for (int r = 0; r < 8; ++r) acc[s][r] = bias;

#pragma unroll
  for (int k0 = 0; k0 < 256; k0 += 32) {
    v16bf bfr = frag_b(w1t, ncol, 256, k0, lane);
#pragma unroll
    for (int s = 0; s < 2; ++s) {
      v16bf a = frag_a(ef + s * 16 * 256, m, 256, k0, lane);
      acc[s] = __builtin_amdgcn_wmma_f32_16x16x32_bf16(false, a, false, bfr,
                                                       (short)0, acc[s],
                                                       false, false);
    }
  }

#pragma unroll
  for (int s = 0; s < 2; ++s) {
#pragma unroll
    for (int r = 0; r < 8; ++r) {
      int rr = s * 16 + r + (lane >> 4) * 8;
      hmid[rr * HID + ncol] = gelu_exact(acc[s][r]);
    }
  }
  __syncthreads();

  if (t < QROWS) {
    int qq = q0 + t;
    if (qq < EQ) {
      float accum = b2[0];
#pragma unroll 8
      for (int k = 0; k < HID; ++k) accum += hmid[t * HID + k] * w2[k];
      out[qq] = 1.f / (1.f + expf(-accum));
    }
  }
}

// ---------------------------------------------------------------------------
// Host launch
// ---------------------------------------------------------------------------
static inline size_t align256(size_t v) { return (v + 255) & ~(size_t)255; }

extern "C" void kernel_launch(void* const* d_in, const int* in_sizes, int n_in,
                              void* d_out, int out_size, void* d_ws,
                              size_t ws_size, hipStream_t stream) {
  const float* x     = (const float*)d_in[0];
  const int*   ei    = (const int*)d_in[1];
  const float* ea    = (const float*)d_in[2];
  const int*   eli   = (const int*)d_in[3];
  const float* relw  = (const float*)d_in[4];
  const float* relb  = (const float*)d_in[5];
  const float* rootw = (const float*)d_in[6];
  const float* bng   = (const float*)d_in[7];
  const float* bnb   = (const float*)d_in[8];
  const float* w1    = (const float*)d_in[9];
  const float* b1    = (const float*)d_in[10];
  const float* w2    = (const float*)d_in[11];
  const float* b2    = (const float*)d_in[12];

  const int Nn = in_sizes[0] / HID;
  const int E  = in_sizes[1] / 2;
  const int EQ = in_sizes[3] / 2;

  const size_t featB = (size_t)Nn * HID * sizeof(float);

  char* ws = (char*)d_ws;
  float* agg   = (float*)ws; ws += align256(featB);
  float* h     = (float*)ws; ws += align256(featB);
  float* xcur  = (float*)ws; ws += align256(featB);
  float* stats = (float*)ws; ws += align256(256 * sizeof(float));
  __bf16* relwt  = (__bf16*)ws; ws += align256((size_t)NCONV * HID * HID * 2);
  __bf16* rootwt = (__bf16*)ws; ws += align256((size_t)NCONV * HID * HID * 2);
  __bf16* w1t    = (__bf16*)ws; ws += align256((size_t)2 * HID * HID * 2);
  (void)ws_size;

  // One-time weight conversion (deterministic, re-run every call).
  prep_weights<<<(NCONV * HID * HID + 255) / 256, 256, 0, stream>>>(
      relw, rootw, w1, relwt, rootwt, w1t);

  // xcur = x
  hipMemcpyAsync(xcur, x, featB, hipMemcpyDeviceToDevice, stream);

  const int ggrid = (Nn + CROWS - 1) / CROWS;

  for (int l = 0; l < NCONV; ++l) {
    zero_f32<<<2048, 256, 0, stream>>>(agg, (size_t)Nn * HID);
    scatter_edges<<<4096, 256, 0, stream>>>(xcur, ei, ea, agg, E);
    conv_gemm_gelu<<<ggrid, 256, 0, stream>>>(
        agg, xcur, relwt + (size_t)l * HID * HID,
        rootwt + (size_t)l * HID * HID, relb + l * HID, h, Nn);
    zero_f32<<<1, 256, 0, stream>>>(stats, 256);
    bn_stats<<<480, 256, 0, stream>>>(h, stats, Nn);
    bn_apply<<<960, 256, 0, stream>>>(h, stats, bng + l * HID, bnb + l * HID,
                                      xcur, Nn);
  }

  edge_mlp<<<(EQ + QROWS - 1) / QROWS, 256, 0, stream>>>(
      xcur, eli, w1t, b1, w2, b2, (float*)d_out, EQ);
}
